// ConditionerForScreener_55765855371538
// MI455X (gfx1250) — matrix operand order repaired and checked by reading, not compiled, not run on hardware
//
#include <hip/hip_runtime.h>
#include <hip/hip_bf16.h>

typedef __attribute__((ext_vector_type(16))) _Float16 v16h;
typedef __attribute__((ext_vector_type(8)))  _Float16 v8h;
typedef __attribute__((ext_vector_type(8)))  float    v8f;
typedef __attribute__((ext_vector_type(4)))  int      v4i;

#define SPATIAL 110592   // 48*48*48
#define CCH 96           // channels
#define KPROT 512        // prototypes
#define NT_TILES 32      // 512/16
#define KC_CHUNKS 3      // 96/32
#define NTILES (KC_CHUNKS * NT_TILES)      // 96 B-tiles
#define BHALF  (NTILES * 512)              // 49152 halves = 96 KB
#define COPY_CHUNKS ((BHALF * 2 / 16) / 256)   // 24 x 16B per thread

#if __has_builtin(__builtin_amdgcn_global_load_async_to_lds_b128)
#define HAVE_ASYNC_BUILTIN 1
#else
#define HAVE_ASYNC_BUILTIN 0
#endif

// ---------------------------------------------------------------------------
// Kernel 1: normalize prototypes; scatter into LDS-friendly B-tile layout.
//   Tile t = kc*32 + nt: half index = part*256 + lane_b*8 + (h&7),
//   lane_b = n + 16*(c_local>=16), h = c_local&15, part = h>>3.
//   => hot loop: two ds_load_b128 per lane at 16B lane stride (minimal
//      conflict), offsets t*1024 + lane*16 and +512.
// ---------------------------------------------------------------------------
__global__ void proto_norm_kernel(const float* __restrict__ protos,
                                  _Float16* __restrict__ pnB) {
    const int k    = blockIdx.x;        // prototype 0..511
    const int lane = threadIdx.x;       // 0..31
    const float* row = protos + (size_t)k * CCH;

    float v0 = row[lane], v1 = row[lane + 32], v2 = row[lane + 64];
    float ss = v0 * v0 + v1 * v1 + v2 * v2;
#pragma unroll
    for (int m = 16; m >= 1; m >>= 1) ss += __shfl_xor(ss, m, 32);
    const float inv = 1.0f / fmaxf(sqrtf(ss), 1e-12f);

    const int nt = k >> 4;
    const int n  = k & 15;
    const float vals[3] = {v0, v1, v2};
#pragma unroll
    for (int j = 0; j < 3; ++j) {
        const int cl     = lane;              // c_local = c & 31 == lane
        const int lane_b = n + (cl & 16);
        const int h      = cl & 15;
        const int t      = j * NT_TILES + nt; // kc = j
        const int part   = h >> 3;
        pnB[t * 512 + part * 256 + lane_b * 8 + (h & 7)] =
            (_Float16)(vals[j] * inv);
    }
}

// ---------------------------------------------------------------------------
// Kernel 2: block = 8 waves x 32 positions = 256 positions.
//   1) async-DMA the 96KB swizzled prototypes into LDS (ASYNCcnt path)
//   2) overlap: load per-wave A-fragments (f32->f16) while the DMA runs
//   3) s_wait_asynccnt 0 + barrier, then nt loop:
//        2 ds_load_b128 + 2 independent WMMA chains per K-chunk
//   4) running per-lane argmax, 16-lane shfl_xor butterfly, tie-break low idx.
// ---------------------------------------------------------------------------
__global__ __launch_bounds__(256)
void sim_argmax_kernel(const float* __restrict__ x,
                       const _Float16* __restrict__ pnB,
                       int* __restrict__ out) {
    __shared__ __align__(16) _Float16 bsh[BHALF];   // 96 KB of 320 KB/WGP

    const int tid = threadIdx.x;

    // --- stage prototypes into LDS: direct memory->LDS async copy ---
    {
        const float4* src = (const float4*)pnB;
        float4*       dst = (float4*)bsh;
#pragma unroll
        for (int i = 0; i < COPY_CHUNKS; ++i) {
            const float4* g = src + i * 256 + tid;
            float4*       l = dst + i * 256 + tid;
#if HAVE_ASYNC_BUILTIN
            __builtin_amdgcn_global_load_async_to_lds_b128(
                (__attribute__((address_space(1))) v4i*)(float4*)g,
                (__attribute__((address_space(3))) v4i*)l,
                /*offset=*/0, /*cpol=*/0);
#else
            const unsigned long long ga = (unsigned long long)g;
            const unsigned la =
                (unsigned)(size_t)(__attribute__((address_space(3))) void*)l;
            asm volatile("global_load_async_to_lds_b128 %0, %1, off"
                         :: "v"(la), "v"(ga) : "memory");
#endif
        }
    }

    // --- A fragments: overlap these HBM loads with the async LDS DMA ---
    const int lane = tid & 31;
    const int wave = tid >> 5;
    const int p0   = blockIdx.x * 256 + wave * 32;   // 2 M-tiles per wave
    const int b    = p0 / SPATIAL;                   // never straddles batch
    const int spb  = p0 - b * SPATIAL;
    const int hi   = (lane >> 4) & 1;

    const float* xb = x + (size_t)b * CCH * SPATIAL + spb + (lane & 15);

    v16h a[2][KC_CHUNKS];
#pragma unroll
    for (int mt = 0; mt < 2; ++mt) {
#pragma unroll
        for (int kc = 0; kc < KC_CHUNKS; ++kc) {
#pragma unroll
            for (int s = 0; s < 16; ++s) {
                const int cl = s + 16 * hi;
                const int c  = kc * 32 + cl;
                a[mt][kc][s] = (_Float16)xb[(size_t)c * SPATIAL + mt * 16];
            }
        }
    }

    // All of this wave's DMAs must land before anyone reads LDS.
#if __has_builtin(__builtin_amdgcn_s_wait_asynccnt)
    __builtin_amdgcn_s_wait_asynccnt(0);
#else
    asm volatile("s_wait_asynccnt 0" ::: "memory");
#endif
    __syncthreads();

    float bestv[2][8];
    int   besti[2][8];
#pragma unroll
    for (int mt = 0; mt < 2; ++mt)
#pragma unroll
        for (int r = 0; r < 8; ++r) { bestv[mt][r] = -3.0e38f; besti[mt][r] = 0; }

    for (int nt = 0; nt < NT_TILES; ++nt) {
        v8f acc0 = {}, acc1 = {};
#pragma unroll
        for (int kc = 0; kc < KC_CHUNKS; ++kc) {
            const int t = kc * NT_TILES + nt;
            const v8h* pl = (const v8h*)(bsh + t * 512);
            const v8h lo  = pl[lane];        // ds_load_b128, lane*16B
            const v8h hi8 = pl[32 + lane];   // ds_load_b128, +512B
            const v16h bf = __builtin_shufflevector(
                lo, hi8, 0, 1, 2, 3, 4, 5, 6, 7, 8, 9, 10, 11, 12, 13, 14, 15);
            acc0 = __builtin_amdgcn_wmma_f32_16x16x32_f16(
                false, a[0][kc], false, bf, (short)0, acc0, false, false);
            acc1 = __builtin_amdgcn_wmma_f32_16x16x32_f16(
                false, a[1][kc], false, bf, (short)0, acc1, false, false);
        }
        const int kidx = nt * 16 + (lane & 15);
#pragma unroll
        for (int r = 0; r < 8; ++r) {
            const float u = acc0[r];
            if (u > bestv[0][r]) { bestv[0][r] = u; besti[0][r] = kidx; }
            const float w = acc1[r];
            if (w > bestv[1][r]) { bestv[1][r] = w; besti[1][r] = kidx; }
        }
    }

    // --- butterfly max-reduce across the 16 lanes sharing each row group ---
#pragma unroll
    for (int mt = 0; mt < 2; ++mt) {
#pragma unroll
        for (int m = 1; m <= 8; m <<= 1) {
#pragma unroll
            for (int r = 0; r < 8; ++r) {
                const float ov = __shfl_xor(bestv[mt][r], m, 32);
                const int   oi = __shfl_xor(besti[mt][r], m, 32);
                if (ov > bestv[mt][r] ||
                    (ov == bestv[mt][r] && oi < besti[mt][r])) {
                    bestv[mt][r] = ov; besti[mt][r] = oi;
                }
            }
        }
    }

    // Lane 0: rows 0..7, lane 16: rows 8..15 (per M-tile).
    if (lane == 0 || lane == 16) {
#pragma unroll
        for (int mt = 0; mt < 2; ++mt) {
            const int base = p0 + mt * 16 + hi * 8;
#pragma unroll
            for (int r = 0; r < 8; ++r) out[base + r] = besti[mt][r];
        }
    }
}

extern "C" void kernel_launch(void* const* d_in, const int* in_sizes, int n_in,
                              void* d_out, int out_size, void* d_ws, size_t ws_size,
                              hipStream_t stream) {
    const float* x      = (const float*)d_in[0];   // [2, 96, 48, 48, 48] f32
    const float* protos = (const float*)d_in[1];   // [512, 96] f32
    int*       out = (int*)d_out;                  // [2, 48, 48, 48] int32 argmax
    _Float16*  pnB = (_Float16*)d_ws;              // 96 KB swizzled f16 prototypes

    proto_norm_kernel<<<KPROT, 32, 0, stream>>>(protos, pnB);

    const int positions = in_sizes[0] / CCH;       // 221184
    const int blocks    = positions / 256;         // 864
    sim_argmax_kernel<<<blocks, 256, 0, stream>>>(x, pnB, out);
}
